// LNO_69913477644331
// MI455X (gfx1250) — matrix operand order repaired and checked by reading, not compiled, not run on hardware
//
#include <hip/hip_runtime.h>

#define NT     1024
#define NV     128
#define NXS    512
#define MDIM   65536          // NV*NXS, contraction length of big GEMM
#define KNUM   257
#define KP     272            // 17*16, stage-2 padded K
#define KP2    288            // 9*32, stage-3 padded K
#define NXO    2048
#define SPLITS 16
#define MSLICE (MDIM / SPLITS) // 4096

typedef __attribute__((ext_vector_type(16))) __bf16 v16bf;
typedef __attribute__((ext_vector_type(8)))  float  v8f;

union Frag {
  v16bf v;
  uint4 q[2];
  unsigned short u[16];
};

__device__ __forceinline__ unsigned short f2bfu(float x) {
  unsigned u = __builtin_bit_cast(unsigned, x);
  u += 0x7FFFu + ((u >> 16) & 1u);   // round-to-nearest-even
  return (unsigned short)(u >> 16);
}

// A-style fragment: 16 bf16 at byte offsets [0,16) and [32,48)  (k = kb..kb+7, kb+16..kb+23)
__device__ __forceinline__ v16bf load_frag_gap(const void* p) {
  Frag f;
  f.q[0] = *(const uint4*)((const char*)p);
  f.q[1] = *(const uint4*)((const char*)p + 32);
  return f.v;
}
// B-style fragment: 32 contiguous bytes (k = kb..kb+15)
__device__ __forceinline__ v16bf load_frag_cont(const void* p) {
  Frag f;
  f.q[0] = *(const uint4*)((const char*)p);
  f.q[1] = *(const uint4*)((const char*)p + 16);
  return f.v;
}
__device__ __forceinline__ v8f wmma_bf16(v16bf a, v16bf b, v8f c) {
  return __builtin_amdgcn_wmma_f32_16x16x32_bf16(false, a, false, b, (short)0, c, false, false);
}
__device__ __forceinline__ v8f vzero8() {
  v8f z = {0.f, 0.f, 0.f, 0.f, 0.f, 0.f, 0.f, 0.f};
  return z;
}

// CDNA5 async global->LDS copy of 16 bytes (GLOBAL_LOAD_ASYNC_TO_LDS_B128).
// ldsAddr: low 32 bits of a generic pointer to __shared__ (== LDS byte offset,
// ISA section 10.2: LDS aperture uses addr[31:0]). Tracked by ASYNCcnt.
__device__ __forceinline__ void async_copy_b128(void* ldsDst, const void* gsrc) {
  const unsigned lds_addr = (unsigned)(size_t)ldsDst;
  const unsigned long long gaddr = (unsigned long long)(size_t)gsrc;
  asm volatile("global_load_async_to_lds_b128 %0, %1, off"
               :: "v"(lds_addr), "v"(gaddr)
               : "memory");
}
__device__ __forceinline__ void wait_asynccnt0() {
  asm volatile("s_wait_asynccnt 0x0" ::: "memory");
}

// ---------------------------------------------------------------------------
// Stage 1: psi MLP. Layers 1-2 scalar (LDS-broadcast weights), layer 3 via
// WMMA: Psi^T tile = (W3^T as A: 16k x 32i) x (h2^T as B: 32i x 16m).
// Writes PsiT[k][m] bf16 (k-major) so stage-2 B fragments are contiguous.
// ---------------------------------------------------------------------------
__global__ __launch_bounds__(256) void psi_kernel(
    const float* __restrict__ xs, const float* __restrict__ vs,
    const float* __restrict__ W1, const float* __restrict__ b1,
    const float* __restrict__ W2, const float* __restrict__ b2,
    const float* __restrict__ W3, const float* __restrict__ b3,
    unsigned short* __restrict__ psiT)
{
  __shared__ float sW12[2272];                         // W1(128) b1(64) W2(2048) b2(32)
  __shared__ float sb3[KP];
  __shared__ __align__(16) unsigned short sW3T[KP * 32];   // [k][i] bf16
  __shared__ __align__(16) unsigned short sh2[256 * 32];   // [m][i] bf16
  const int tid = threadIdx.x;

  for (int i = tid; i < 128;  i += 256) sW12[i]        = W1[i];
  for (int i = tid; i < 64;   i += 256) sW12[128 + i]  = b1[i];
  for (int i = tid; i < 2048; i += 256) sW12[192 + i]  = W2[i];
  for (int i = tid; i < 32;   i += 256) sW12[2240 + i] = b2[i];
  for (int i = tid; i < KP;   i += 256) sb3[i] = (i < KNUM) ? b3[i] : 0.f;
  for (int i = tid; i < KP * 32; i += 256) {
    const int k = i >> 5, j = i & 31;
    sW3T[i] = (k < KNUM) ? f2bfu(W3[j * KNUM + k]) : (unsigned short)0;
  }
  __syncthreads();

  const int p = blockIdx.x * 256 + tid;
  const float x = xs[p & (NXS - 1)];
  const float v = vs[p >> 9];
  float h1[64];
#pragma unroll
  for (int j = 0; j < 64; ++j)
    h1[j] = fmaxf(0.f, fmaf(x, sW12[j], fmaf(v, sW12[64 + j], sW12[128 + j])));
  for (int i = 0; i < 32; ++i) {
    float acc = sW12[2240 + i];
#pragma unroll
    for (int j = 0; j < 64; ++j) acc = fmaf(h1[j], sW12[192 + j * 32 + i], acc);
    sh2[tid * 32 + i] = f2bfu(fmaxf(0.f, acc));
  }
  __syncthreads();

  const int w = tid >> 5, lane = tid & 31;
  const int half8  = (lane < 16) ? 0 : 8;
  const int half16 = (lane < 16) ? 0 : 16;
  const int pgBase = blockIdx.x * 256;
  for (int idx = w; idx < 17 * 16; idx += 8) {   // 34 tiles per wave, uniform
    const int kt = idx % 17, mt = idx / 17;
    v16bf a = load_frag_gap ((const char*)sW3T + (kt * 16 + (lane & 15)) * 64 + half8 * 2);
    v16bf b = load_frag_gap ((const char*)sh2  + (mt * 16 + (lane & 15)) * 64 + half8 * 2);
    v8f c = vzero8();
    c = wmma_bf16(a, b, c);
    const int pg = pgBase + mt * 16 + (lane & 15);
#pragma unroll
    for (int r = 0; r < 8; ++r) {
      const int kg = kt * 16 + r + half8;
      psiT[(size_t)kg * MDIM + pg] = f2bfu(c[r] + sb3[kg]);
    }
  }
}

// ---------------------------------------------------------------------------
// Stage 2: weights = f (1024 x 65536, fp32->bf16 on the fly) @ Psi (65536 x 272)
// Block: 8 waves = 128 t-rows, all 17 k-tiles, one of 16 m-splits.
// B (Psi slice, 32 x 272) double-buffered in LDS via GLOBAL_LOAD_ASYNC_TO_LDS,
// overlapped with the WMMA stream. Split-K partials to workspace.
// ---------------------------------------------------------------------------
__global__ __launch_bounds__(256) void gemm1_kernel(
    const float* __restrict__ f, const unsigned short* __restrict__ psiT,
    float* __restrict__ partials)
{
  __shared__ __align__(16) unsigned short sB[2][KP * 32];  // [buf][k][32 m] bf16
  const int tid = threadIdx.x;
  const int w = tid >> 5, lane = tid & 31;
  const int tb = blockIdx.x & 7;
  const int s  = blockIdx.x >> 3;
  const int rowBase = tb * 128 + w * 16;
  const int row = rowBase + (lane & 15);
  const float* fr = f + (size_t)row * MDIM;
  const int m0base = s * MSLICE;

  v8f acc[17];
#pragma unroll
  for (int i = 0; i < 17; ++i) acc[i] = vzero8();

  // prologue: async-stage buffer 0
  for (int c = tid; c < KP * 4; c += 256) {
    const int k = c >> 2, part = c & 3;
    async_copy_b128(&sB[0][k * 32 + part * 8],
                    psiT + (size_t)k * MDIM + m0base + part * 8);
  }

  const int kb = (lane < 16) ? 0 : 8;
  const int mb = (lane < 16) ? 0 : 16;
  const int NSTEP = MSLICE / 32;   // 128
  for (int step = 0; step < NSTEP; ++step) {
    wait_asynccnt0();              // my async writes into current buffer done
    __syncthreads();               // everyone's staging visible, prev reads done
    const int cur = step & 1;
    if (step + 1 < NSTEP) {
      const int m0n = m0base + (step + 1) * 32;
      for (int c = tid; c < KP * 4; c += 256) {
        const int k = c >> 2, part = c & 3;
        async_copy_b128(&sB[cur ^ 1][k * 32 + part * 8],
                        psiT + (size_t)k * MDIM + m0n + part * 8);
      }
    }
    const int m0 = m0base + step * 32;
    const float* ap = fr + m0 + kb;
    const float4 a0 = *(const float4*)(ap);
    const float4 a1 = *(const float4*)(ap + 4);
    const float4 a2 = *(const float4*)(ap + 16);
    const float4 a3 = *(const float4*)(ap + 20);
    if (step + 8 < NSTEP) __builtin_prefetch(fr + m0 + 8 * 32, 0, 0);
    Frag af;
    af.u[0]  = f2bfu(a0.x); af.u[1]  = f2bfu(a0.y); af.u[2]  = f2bfu(a0.z); af.u[3]  = f2bfu(a0.w);
    af.u[4]  = f2bfu(a1.x); af.u[5]  = f2bfu(a1.y); af.u[6]  = f2bfu(a1.z); af.u[7]  = f2bfu(a1.w);
    af.u[8]  = f2bfu(a2.x); af.u[9]  = f2bfu(a2.y); af.u[10] = f2bfu(a2.z); af.u[11] = f2bfu(a2.w);
    af.u[12] = f2bfu(a3.x); af.u[13] = f2bfu(a3.y); af.u[14] = f2bfu(a3.z); af.u[15] = f2bfu(a3.w);
#pragma unroll
    for (int kt = 0; kt < 17; ++kt) {
      v16bf b = load_frag_gap ((const char*)sB[cur] + (kt * 16 + (lane & 15)) * 64 + kb * 2);
      acc[kt] = wmma_bf16(af.v, b, acc[kt]);
    }
  }

  float* pbase = partials + (size_t)s * NT * KP;
#pragma unroll
  for (int kt = 0; kt < 17; ++kt) {
    const int col = kt * 16 + (lane & 15);
#pragma unroll
    for (int r = 0; r < 8; ++r) {
      const int rg = rowBase + r + ((lane < 16) ? 0 : 8);
      pbase[(size_t)rg * KP + col] = acc[kt][r];
    }
  }
}

// ---------------------------------------------------------------------------
// Stage 2b: reduce split-K partials, scale by hx*hv, emit bf16 weights (pad 288)
// ---------------------------------------------------------------------------
__global__ __launch_bounds__(256) void reduce_kernel(
    const float* __restrict__ partials, const float* __restrict__ xs,
    const float* __restrict__ vs, unsigned short* __restrict__ wbf)
{
  const int i = blockIdx.x * 256 + threadIdx.x;
  if (i >= NT * KP2) return;
  const int t = i / KP2, k = i - t * KP2;
  float val = 0.f;
  if (k < KP) {
    const float scale = (xs[1] - xs[0]) * (vs[1] - vs[0]);
    float ssum = 0.f;
#pragma unroll
    for (int sp = 0; sp < SPLITS; ++sp)
      ssum += partials[(size_t)sp * NT * KP + (size_t)t * KP + k];
    val = ssum * scale;
  }
  wbf[i] = f2bfu(val);
}

// ---------------------------------------------------------------------------
// Stage 3a: phi^T[x_out][k] in bf16 (rows 0..127 sin((k+1)wx), 128..256 cos)
// ---------------------------------------------------------------------------
__global__ __launch_bounds__(256) void phi_kernel(
    const float* __restrict__ x_out, const float* __restrict__ xs,
    unsigned short* __restrict__ phibT)
{
  const int i = blockIdx.x * 256 + threadIdx.x;
  if (i >= NXO * KP2) return;
  const int xo = i / KP2, k = i - xo * KP2;
  const float w = 6.283185307179586f / (xs[NXS - 1] - xs[0]);
  const float xv = x_out[xo];
  float val = 0.f;
  if (k < 128)       val = __sinf((float)(k + 1) * w * xv);
  else if (k < KNUM) val = __cosf((float)(k - 128) * w * xv);
  phibT[i] = f2bfu(val);
}

// ---------------------------------------------------------------------------
// Stage 3b: out(1024x2048) = weights(1024x288 bf16) @ phi(288x2048 bf16), f32 out
// One 16x16 tile per wave, 9 WMMA steps.
// ---------------------------------------------------------------------------
__global__ __launch_bounds__(256) void gemm2_kernel(
    const unsigned short* __restrict__ wbf, const unsigned short* __restrict__ phibT,
    float* __restrict__ out)
{
  const int tid = threadIdx.x;
  const int w = tid >> 5, lane = tid & 31;
  const int idx = blockIdx.x * 8 + w;
  const int tt = idx >> 7;      // 128 xo-tiles per t-tile row
  const int xt = idx & 127;
  const int row = tt * 16 + (lane & 15);
  const int xo  = xt * 16 + (lane & 15);
  const int kb = (lane < 16) ? 0 : 8;
  const int mb = (lane < 16) ? 0 : 16;
  v8f c = vzero8();
#pragma unroll
  for (int stp = 0; stp < KP2 / 32; ++stp) {
    const int ks = stp * 32;
    v16bf a = load_frag_gap ((const char*)wbf   + ((size_t)row * KP2 + ks + kb) * 2);
    v16bf b = load_frag_gap ((const char*)phibT + ((size_t)xo  * KP2 + ks + kb) * 2);
    c = wmma_bf16(a, b, c);
  }
#pragma unroll
  for (int r = 0; r < 8; ++r) {
    const int rg = tt * 16 + r + ((lane < 16) ? 0 : 8);
    out[(size_t)rg * NXO + xt * 16 + (lane & 15)] = c[r];
  }
}

// ---------------------------------------------------------------------------
extern "C" void kernel_launch(void* const* d_in, const int* in_sizes, int n_in,
                              void* d_out, int out_size, void* d_ws, size_t ws_size,
                              hipStream_t stream) {
  (void)in_sizes; (void)n_in; (void)out_size; (void)ws_size;
  const float* x_out = (const float*)d_in[0];
  const float* f     = (const float*)d_in[1];
  const float* xs    = (const float*)d_in[2];
  const float* vs    = (const float*)d_in[3];
  const float* W1    = (const float*)d_in[4];
  const float* b1    = (const float*)d_in[5];
  const float* W2    = (const float*)d_in[6];
  const float* b2    = (const float*)d_in[7];
  const float* W3    = (const float*)d_in[8];
  const float* b3    = (const float*)d_in[9];
  float* out = (float*)d_out;

  char* ws = (char*)d_ws;
  size_t off = 0;
  unsigned short* psiT     = (unsigned short*)(ws + off); off += (size_t)KP * MDIM * 2;        // 35.65 MB
  float*          partials = (float*)         (ws + off); off += (size_t)SPLITS * NT * KP * 4; // 17.83 MB
  unsigned short* wbf      = (unsigned short*)(ws + off); off += (size_t)NT * KP2 * 2;         // 0.59 MB
  unsigned short* phibT    = (unsigned short*)(ws + off); off += (size_t)NXO * KP2 * 2;        // 1.18 MB

  psi_kernel   <<<MDIM / 256, 256, 0, stream>>>(xs, vs, W1, b1, W2, b2, W3, b3, psiT);
  gemm1_kernel <<<8 * SPLITS, 256, 0, stream>>>(f, psiT, partials);
  reduce_kernel<<<(NT * KP2) / 256, 256, 0, stream>>>(partials, xs, vs, wbf);
  phi_kernel   <<<(NXO * KP2) / 256, 256, 0, stream>>>(x_out, xs, phibT);
  gemm2_kernel <<<(NT / 16) * (NXO / 16) / 8, 256, 0, stream>>>(wbf, phibT, out);
}
